// GATNet_28630251995358
// MI455X (gfx1250) — matrix operand order repaired and checked by reading, compiled, not run on hardware
//
#include <hip/hip_runtime.h>
#include <hip/hip_bf16.h>

#define N_NODES   100000
#define N_EDGES   1600000
#define HID       128
#define HEADS1    8
#define CH1       16
#define REPR      64
#define LRELU     0.2f
#define SOFT_EPS  1e-16f
#define NEG_FLT_MAX -3.402823466e38f
#define LDS_STRIDE (HID + 4)   // pad: bank = (4*row + k) % 64, conflict-free

typedef float  v2f   __attribute__((ext_vector_type(2)));
typedef float  v8f   __attribute__((ext_vector_type(8)));
typedef __bf16 v16bf __attribute__((ext_vector_type(16)));

#if __has_builtin(__builtin_amdgcn_wmma_f32_16x16x4_f32)
#define HAVE_WMMA_F32 1
#else
#define HAVE_WMMA_F32 0
#endif

// ---------------------------------------------------------------------------
// GEMM: C[M,N] = A[M,128] * B[128,N], row-major fp32, K fixed at 128.
// One block (8 waves) per 16-row tile: stage A tile in LDS once (coalesced
// float4), then one wave per 16-col tile runs the f32 WMMA K-loop from LDS.
// ---------------------------------------------------------------------------
__global__ __launch_bounds__(256) void gemm_wmma_kernel(
    const float* __restrict__ A, const float* __restrict__ B,
    float* __restrict__ C, int N)
{
    __shared__ __align__(16) float As[16 * LDS_STRIDE];

    const int wave = threadIdx.x >> 5;
    const int lane = threadIdx.x & 31;
    const int row0 = blockIdx.x << 4;

    // stage 16 x 128 tile of A: 512 float4s across 256 threads (2 each)
    const float4* Ag = (const float4*)(A + (size_t)row0 * HID);
    for (int i = threadIdx.x; i < (16 * HID) / 4; i += 256) {
        float4 v = Ag[i];
        const int r = i >> 5;            // (i*4) / 128
        const int c = (i & 31) << 2;     // (i*4) % 128
        *(float4*)&As[r * LDS_STRIDE + c] = v;
    }
    __syncthreads();

    const int cTiles = N >> 4;
    if (wave < cTiles) {                 // wave-uniform: EXEC stays full
        const int col0  = wave << 4;
        const int mrow  = lane & 15;
        const int khalf = lane >> 4;     // 0: K pair {0,1}, 1: {2,3}

        v8f acc = {};
#if HAVE_WMMA_F32
        for (int k0 = 0; k0 < HID; k0 += 4) {
            const int ka = k0 + (khalf << 1);
            v2f a = *(const v2f*)&As[mrow * LDS_STRIDE + ka];  // ds_load_b64
            v2f b;
            b.x = B[(size_t)(ka)     * N + col0 + mrow];
            b.y = B[(size_t)(ka + 1) * N + col0 + mrow];
            acc = __builtin_amdgcn_wmma_f32_16x16x4_f32(
                false, a, false, b, (short)0, acc, false, false);
        }
#else
        for (int k0 = 0; k0 < HID; k0 += 32) {
            v16bf a, b;
            for (int i = 0; i < 16; ++i) {
                const int grp = i >> 3, win = i & 7;
                a[i] = (__bf16)As[mrow * LDS_STRIDE + k0 + grp * 16 + khalf * 8 + win];
                b[i] = (__bf16)B[(size_t)(k0 + grp * 16 + khalf * 8 + win) * N + col0 + mrow];
            }
            acc = __builtin_amdgcn_wmma_f32_16x16x32_bf16(
                false, a, false, b, (short)0, acc, false, false);
        }
#endif
        for (int v = 0; v < 8; ++v)
            C[(size_t)(row0 + v + khalf * 8) * N + col0 + mrow] = acc[v];
    }
}

// ---------------------------------------------------------------------------
// Helpers / elementwise kernels
// ---------------------------------------------------------------------------
__global__ void fill_kernel(float* __restrict__ p, float v, int n)
{
    int i = blockIdx.x * blockDim.x + threadIdx.x;
    if (i < n) p[i] = v;
}

__device__ __forceinline__ void atomicMaxFloat(float* addr, float val)
{
    if (val >= 0.0f) atomicMax((int*)addr, __float_as_int(val));
    else             atomicMin((unsigned int*)addr, __float_as_uint(val));
}

// alpha_s[n,h] = sum_c h[n,h,c]*a_src[h,c] ; same for alpha_d
__global__ void node_alpha_kernel(const float* __restrict__ h,
    const float* __restrict__ a_src, const float* __restrict__ a_dst,
    float* __restrict__ as, float* __restrict__ ad, int n, int H, int C)
{
    int idx = blockIdx.x * blockDim.x + threadIdx.x;
    if (idx >= n * H) return;
    int node = idx / H, hh = idx - node * H;
    const float* hp = h + (size_t)node * H * C + hh * C;
    float s = 0.f, d = 0.f;
    for (int c = 0; c < C; ++c) {
        s += hp[c] * a_src[hh * C + c];
        d += hp[c] * a_dst[hh * C + c];
    }
    as[idx] = s;
    ad[idx] = d;
}

__device__ __forceinline__ void edge_endpoints(
    const int* __restrict__ src, const int* __restrict__ dst,
    int e, int E, int& s, int& d)
{
    if (e < E) { s = src[e]; d = dst[e]; }
    else       { s = e - E;  d = s; }      // self-loops appended
}

// pass A: segment max of leakyrelu(as[src]+ad[dst]) per (dst,h)
__global__ void edge_max_kernel(const int* __restrict__ src, const int* __restrict__ dst,
    int E, int n, const float* __restrict__ as, const float* __restrict__ ad,
    float* __restrict__ fmax, int H)
{
    int idx = blockIdx.x * blockDim.x + threadIdx.x;
    int total = (E + n) * H;
    if (idx >= total) return;
    int e = idx / H, hh = idx - e * H;
    int s, d; edge_endpoints(src, dst, e, E, s, d);
    float v = as[s * H + hh] + ad[d * H + hh];
    v = v > 0.0f ? v : LRELU * v;
    atomicMaxFloat(&fmax[d * H + hh], v);
}

// pass B: ex = exp(e - max); denom[dst,h] += ex; stash ex per (edge,head)
__global__ void edge_sum_kernel(const int* __restrict__ src, const int* __restrict__ dst,
    int E, int n, const float* __restrict__ as, const float* __restrict__ ad,
    const float* __restrict__ fmax, float* __restrict__ denom,
    float* __restrict__ exbuf, int H)
{
    int idx = blockIdx.x * blockDim.x + threadIdx.x;
    int total = (E + n) * H;
    if (idx >= total) return;
    int e = idx / H, hh = idx - e * H;
    int s, d; edge_endpoints(src, dst, e, E, s, d);
    float v = as[s * H + hh] + ad[d * H + hh];
    v = v > 0.0f ? v : LRELU * v;
    float ex = expf(v - fmax[d * H + hh]);
    exbuf[idx] = ex;
    atomicAdd(&denom[d * H + hh], ex);
}

// pass C: agg[dst, ch..ch+3] += h[src, ch..ch+3] * alpha.  4 channels/thread,
// one b128 feature load + 4 native global_atomic_add_f32 per thread.
__global__ void edge_scatter_kernel(const int* __restrict__ src, const int* __restrict__ dst,
    int E, int n, const float* __restrict__ h, const float* __restrict__ exbuf,
    const float* __restrict__ denom, float* __restrict__ agg, int H, int C)
{
    const int HC = H * C;
    const int Q  = HC >> 2;                 // float4 groups per edge
    long long idx = (long long)blockIdx.x * blockDim.x + threadIdx.x;
    long long total = (long long)(E + n) * Q;
    if (idx >= total) return;
    int e  = (int)(idx / Q);
    int ch = (int)(idx - (long long)e * Q) << 2;
    int hh = ch / C;
    int s, d; edge_endpoints(src, dst, e, E, s, d);
    float alpha = exbuf[e * H + hh] / (denom[d * H + hh] + SOFT_EPS);
    const float4 hv = *(const float4*)&h[(size_t)s * HC + ch];
    float* ap = &agg[(size_t)d * HC + ch];
    atomicAdd(ap + 0, hv.x * alpha);
    atomicAdd(ap + 1, hv.y * alpha);
    atomicAdd(ap + 2, hv.z * alpha);
    atomicAdd(ap + 3, hv.w * alpha);
}

// x = elu(x + b[i % dim])
__global__ void elu_bias_kernel(float* __restrict__ x, const float* __restrict__ b,
                                int n, int dim)
{
    int i = blockIdx.x * blockDim.x + threadIdx.x;
    if (i >= n) return;
    float v = x[i] + b[i % dim];
    x[i] = v > 0.0f ? v : (expf(v) - 1.0f);
}

// x += b[i % dim]
__global__ void add_bias_kernel(float* __restrict__ x, const float* __restrict__ b,
                                int n, int dim)
{
    int i = blockIdx.x * blockDim.x + threadIdx.x;
    if (i >= n) return;
    x[i] += b[i % dim];
}

// ---------------------------------------------------------------------------
// Launch
// ---------------------------------------------------------------------------
static inline int blocks_for(long long n) { return (int)((n + 255) / 256); }

extern "C" void kernel_launch(void* const* d_in, const int* in_sizes, int n_in,
                              void* d_out, int out_size, void* d_ws, size_t ws_size,
                              hipStream_t stream)
{
    const int*   ei   = (const int*)d_in[0];            // [2, E]
    const float* emb  = (const float*)d_in[1];          // [N, 128]
    const float* W1   = (const float*)d_in[2];          // [128, 128]
    const float* a_s1 = (const float*)d_in[3];          // [8, 16]
    const float* a_d1 = (const float*)d_in[4];
    const float* b1   = (const float*)d_in[5];          // [128]
    const float* W2   = (const float*)d_in[6];          // [128, 64]
    const float* a_s2 = (const float*)d_in[7];          // [1, 64]
    const float* a_d2 = (const float*)d_in[8];
    const float* b2   = (const float*)d_in[9];          // [64]
    float* out = (float*)d_out;                         // [N, 64]

    const int* src = ei;
    const int* dst = ei + N_EDGES;

    const int Etot = N_EDGES + N_NODES;                 // with self-loops

    // workspace carve (floats; all offsets stay 16B-aligned)
    float* ws   = (float*)d_ws;
    float* h1   = ws;  ws += (size_t)N_NODES * HID;     // 12.8M
    float* agg1 = ws;  ws += (size_t)N_NODES * HID;     // 12.8M (becomes x1 after ELU)
    float* as1  = ws;  ws += (size_t)N_NODES * HEADS1;
    float* ad1  = ws;  ws += (size_t)N_NODES * HEADS1;
    float* mx1  = ws;  ws += (size_t)N_NODES * HEADS1;
    float* dn1  = ws;  ws += (size_t)N_NODES * HEADS1;
    float* ex1  = ws;  ws += (size_t)Etot * HEADS1;     // 13.6M
    float* h2   = ws;  ws += (size_t)N_NODES * REPR;    // 6.4M
    float* as2  = ws;  ws += (size_t)N_NODES;
    float* ad2  = ws;  ws += (size_t)N_NODES;
    float* mx2  = ws;  ws += (size_t)N_NODES;
    float* dn2  = ws;  ws += (size_t)N_NODES;
    float* ex2  = ws;  ws += (size_t)Etot;              // 1.7M

    const int mTile = N_NODES / 16;                     // 6250, exact

    // ---- init ----
    fill_kernel<<<blocks_for((long long)N_NODES * HID), 256, 0, stream>>>(agg1, 0.f, N_NODES * HID);
    fill_kernel<<<blocks_for((long long)N_NODES * HEADS1), 256, 0, stream>>>(mx1, NEG_FLT_MAX, N_NODES * HEADS1);
    fill_kernel<<<blocks_for((long long)N_NODES * HEADS1), 256, 0, stream>>>(dn1, 0.f, N_NODES * HEADS1);
    fill_kernel<<<blocks_for((long long)N_NODES * REPR), 256, 0, stream>>>(out, 0.f, N_NODES * REPR);
    fill_kernel<<<blocks_for(N_NODES), 256, 0, stream>>>(mx2, NEG_FLT_MAX, N_NODES);
    fill_kernel<<<blocks_for(N_NODES), 256, 0, stream>>>(dn2, 0.f, N_NODES);

    // ---- layer 1: GATConv(128 -> 8 x 16) ----
    gemm_wmma_kernel<<<mTile, 256, 0, stream>>>(emb, W1, h1, HID);
    node_alpha_kernel<<<blocks_for((long long)N_NODES * HEADS1), 256, 0, stream>>>(
        h1, a_s1, a_d1, as1, ad1, N_NODES, HEADS1, CH1);
    edge_max_kernel<<<blocks_for((long long)Etot * HEADS1), 256, 0, stream>>>(
        src, dst, N_EDGES, N_NODES, as1, ad1, mx1, HEADS1);
    edge_sum_kernel<<<blocks_for((long long)Etot * HEADS1), 256, 0, stream>>>(
        src, dst, N_EDGES, N_NODES, as1, ad1, mx1, dn1, ex1, HEADS1);
    edge_scatter_kernel<<<blocks_for((long long)Etot * (HID / 4)), 256, 0, stream>>>(
        src, dst, N_EDGES, N_NODES, h1, ex1, dn1, agg1, HEADS1, CH1);
    elu_bias_kernel<<<blocks_for((long long)N_NODES * HID), 256, 0, stream>>>(
        agg1, b1, N_NODES * HID, HID);

    // ---- layer 2: GATConv(128 -> 1 x 64) ----
    gemm_wmma_kernel<<<mTile, 256, 0, stream>>>(agg1, W2, h2, REPR);
    node_alpha_kernel<<<blocks_for(N_NODES), 256, 0, stream>>>(
        h2, a_s2, a_d2, as2, ad2, N_NODES, 1, REPR);
    edge_max_kernel<<<blocks_for(Etot), 256, 0, stream>>>(
        src, dst, N_EDGES, N_NODES, as2, ad2, mx2, 1);
    edge_sum_kernel<<<blocks_for(Etot), 256, 0, stream>>>(
        src, dst, N_EDGES, N_NODES, as2, ad2, mx2, dn2, ex2, 1);
    edge_scatter_kernel<<<blocks_for((long long)Etot * (REPR / 4)), 256, 0, stream>>>(
        src, dst, N_EDGES, N_NODES, h2, ex2, dn2, out, 1, REPR);
    add_bias_kernel<<<blocks_for((long long)N_NODES * REPR), 256, 0, stream>>>(
        out, b2, N_NODES * REPR, REPR);
}
